// ChildSumTreeLSTMCell_9380208575287
// MI455X (gfx1250) — compile-verified, matrix operand order checked by reading, not run on hardware
//
#include <hip/hip_runtime.h>
#include <hip/hip_bf16.h>
#include <math.h>

#define NN 200000
#define NE 200000
#define HS 256

typedef __attribute__((ext_vector_type(16))) __bf16        v16bf;
typedef __attribute__((ext_vector_type(8)))  float         v8f;
typedef __attribute__((ext_vector_type(4)))  unsigned int  v4u;
typedef __attribute__((ext_vector_type(8)))  unsigned int  v8u;

__device__ __forceinline__ unsigned short f2bf(float f) {
    unsigned int u = __float_as_uint(f);
    u += 0x7FFFu + ((u >> 16) & 1u);          // round-to-nearest-even
    return (unsigned short)(u >> 16);
}

// Fast gates on the TRANS unit: v_exp_f32 + v_rcp_f32, no IEEE division chain.
__device__ __forceinline__ float fast_sigmoid(float x) {
    return __builtin_amdgcn_rcpf(1.0f + __expf(-x));
}
__device__ __forceinline__ float fast_tanh(float x) {
    // tanh(x) = 1 - 2/(1 + exp(2x))
    return 1.0f - 2.0f * __builtin_amdgcn_rcpf(1.0f + __expf(2.0f * x));
}

// Load one bf16 A/B WMMA fragment (16x32 / 32x16) for this lane from a
// row-major row pointer: two contiguous 16-byte chunks per lane.
// half = lane>>4 selects K 0-7,16-23 (half=0) vs K 8-15,24-31 (half=1).
__device__ __forceinline__ v16bf load_frag(const unsigned short* row, int kb, int half) {
    v8u r;
    r.lo = *(const v4u*)(row + kb + half * 8);
    r.hi = *(const v4u*)(row + kb + 16 + half * 8);
    return __builtin_bit_cast(v16bf, r);
}

// ---------------- init kernels ----------------

__global__ void zero_ws_kernel(float4* __restrict__ p, long n4) {
    long i = (long)blockIdx.x * blockDim.x + threadIdx.x;
    long stride = (long)gridDim.x * blockDim.x;
    float4 z = make_float4(0.f, 0.f, 0.f, 0.f);
    for (; i < n4; i += stride) p[i] = z;
}

__global__ void convert_weights_kernel(const float* __restrict__ Ufw_f,
                                       const float* __restrict__ Wiou,
                                       const float* __restrict__ Uiou,
                                       unsigned short* __restrict__ UfwB,
                                       unsigned short* __restrict__ WcatB) {
    int i = blockIdx.x * blockDim.x + threadIdx.x;
    if (i < 256 * 256) UfwB[i] = f2bf(Ufw_f[i]);
    if (i < 768 * 512) {
        int n = i >> 9, k = i & 511;
        float v = (k < 256) ? Wiou[n * 256 + k] : Uiou[n * 256 + (k - 256)];
        WcatB[i] = f2bf(v);
    }
}

// ---------------- edge kernel ----------------
// 16 edges per block, 128 threads = 4 waves, each wave -> 4 of 16 N-tiles.
// f = sigmoid(h[src] @ Ufw^T + b); atomicAdd f*c[src] -> c_agg[dst];
// atomicAdd h[src] -> h_tild[dst] during staging.
__global__ __launch_bounds__(128) void edge_kernel(
    const float* __restrict__ h, const float* __restrict__ c,
    const int* __restrict__ esrc, const int* __restrict__ edst,
    const unsigned short* __restrict__ Ufw, const float* __restrict__ Ufb,
    float* __restrict__ h_tild, float* __restrict__ c_agg)
{
    __shared__ unsigned short hA[16][280];   // bf16 A tile, padded stride
    __shared__ float          cS[16][256];   // f32 c_child tile
    __shared__ int sS[16], sD[16];

    const int tid  = threadIdx.x;
    const int tile = blockIdx.x;

    if (tid < 16) { sS[tid] = esrc[tile * 16 + tid]; sD[tid] = edst[tile * 16 + tid]; }
    __syncthreads();

    // stage 16 rows (float4 granularity), convert h->bf16, scatter-add h->h_tild
    for (int i = tid; i < 16 * 64; i += 128) {
        int row = i >> 6, c4 = (i & 63) << 2;
        int src = sS[row], dst = sD[row];
        float4 hv = *(const float4*)&h[(long)src * HS + c4];
        float4 cv = *(const float4*)&c[(long)src * HS + c4];
        hA[row][c4 + 0] = f2bf(hv.x);
        hA[row][c4 + 1] = f2bf(hv.y);
        hA[row][c4 + 2] = f2bf(hv.z);
        hA[row][c4 + 3] = f2bf(hv.w);
        *(float4*)&cS[row][c4] = cv;
        float* ht = &h_tild[(long)dst * HS + c4];
        atomicAdd(ht + 0, hv.x);
        atomicAdd(ht + 1, hv.y);
        atomicAdd(ht + 2, hv.z);
        atomicAdd(ht + 3, hv.w);
    }
    __syncthreads();

    const int lane = tid & 31, wave = tid >> 5;
    const int half = lane >> 4, nq = lane & 15;

    // A fragments (shared across all N-tiles of this wave)
    v16bf afrag[8];
    const unsigned short* arow = &hA[nq][0];
#pragma unroll
    for (int ks = 0; ks < 8; ++ks) afrag[ks] = load_frag(arow, ks * 32, half);

#pragma unroll
    for (int t = 0; t < 4; ++t) {
        int nt = wave * 4 + t;
        int n  = nt * 16 + nq;
        const unsigned short* brow = Ufw + (long)n * HS;
        v8f acc = {};
#pragma unroll
        for (int ks = 0; ks < 8; ++ks) {
            v16bf b = load_frag(brow, ks * 32, half);
            acc = __builtin_amdgcn_wmma_f32_16x16x32_bf16(
                false, afrag[ks], false, b, (short)0, acc, false, false);
        }
        float bias = Ufb[n];
#pragma unroll
        for (int r = 0; r < 8; ++r) {
            int m = r + half * 8;                     // C/D layout: M = r + 8*(lane/16)
            float fg = fast_sigmoid(acc[r] + bias);
            atomicAdd(&c_agg[(long)sD[m] * HS + n], fg * cS[m][n]);
        }
    }
}

// ---------------- node kernel ----------------
// 16 nodes per block, 256 threads = 8 waves, each wave -> 6 of 48 N-tiles.
// iou = [x | h_tild] @ [W_iou ; U_iou]^T + b; then gates + outputs.
__global__ __launch_bounds__(256) void node_kernel(
    const float* __restrict__ x, const float* __restrict__ h_tild,
    const float* __restrict__ c_agg, const unsigned short* __restrict__ Wcat,
    const float* __restrict__ b_iou,
    float* __restrict__ out_h, float* __restrict__ out_c)
{
    __shared__ unsigned short aS[16][520];   // bf16 [x | h_tild] tile
    __shared__ float          iouS[16][776]; // f32 iou tile

    const int tid  = threadIdx.x;
    const int tile = blockIdx.x;

    for (int i = tid; i < 16 * 128; i += 256) {
        int row = i >> 7, c4 = (i & 127) << 2;
        int node = tile * 16 + row;
        float4 v = (c4 < 256) ? *(const float4*)&x[(long)node * HS + c4]
                              : *(const float4*)&h_tild[(long)node * HS + (c4 - 256)];
        aS[row][c4 + 0] = f2bf(v.x);
        aS[row][c4 + 1] = f2bf(v.y);
        aS[row][c4 + 2] = f2bf(v.z);
        aS[row][c4 + 3] = f2bf(v.w);
    }
    __syncthreads();

    const int lane = tid & 31, wave = tid >> 5;
    const int half = lane >> 4, nq = lane & 15;

    v16bf afrag[16];
    const unsigned short* arow = &aS[nq][0];
#pragma unroll
    for (int ks = 0; ks < 16; ++ks) afrag[ks] = load_frag(arow, ks * 32, half);

    for (int t = 0; t < 6; ++t) {
        int nt = wave * 6 + t;
        int n  = nt * 16 + nq;
        const unsigned short* brow = Wcat + (long)n * 512;
        v8f acc = {};
#pragma unroll
        for (int ks = 0; ks < 16; ++ks) {
            v16bf b = load_frag(brow, ks * 32, half);
            acc = __builtin_amdgcn_wmma_f32_16x16x32_bf16(
                false, afrag[ks], false, b, (short)0, acc, false, false);
        }
#pragma unroll
        for (int r = 0; r < 8; ++r) iouS[r + half * 8][n] = acc[r];
    }
    __syncthreads();

    // fused gate epilogue
    for (int i = tid; i < 16 * 256; i += 256) {
        int m = i >> 8, hc = i & 255;
        int node = tile * 16 + m;
        float ig = fast_sigmoid(iouS[m][hc]       + b_iou[hc]);
        float og = fast_sigmoid(iouS[m][256 + hc] + b_iou[256 + hc]);
        float ug = fast_tanh   (iouS[m][512 + hc] + b_iou[512 + hc]);
        float cn = ig * ug + c_agg[(long)node * HS + hc];
        float hn = og * fast_tanh(cn);
        out_h[(long)node * HS + hc] = hn;
        out_c[(long)node * HS + hc] = cn;
    }
}

// ---------------- launch ----------------

extern "C" void kernel_launch(void* const* d_in, const int* in_sizes, int n_in,
                              void* d_out, int out_size, void* d_ws, size_t ws_size,
                              hipStream_t stream) {
    const float* x     = (const float*)d_in[0];
    const float* h     = (const float*)d_in[1];
    const float* c     = (const float*)d_in[2];
    const float* W_iou = (const float*)d_in[3];
    const float* U_iou = (const float*)d_in[4];
    const float* b_iou = (const float*)d_in[5];
    const float* U_f_w = (const float*)d_in[6];
    const float* U_f_b = (const float*)d_in[7];
    const int*   esrc  = (const int*)d_in[8];
    const int*   edst  = (const int*)d_in[9];

    char* ws = (char*)d_ws;
    size_t off = 0;
    float* h_tild = (float*)(ws + off); off += (size_t)NN * HS * sizeof(float);
    float* c_agg  = (float*)(ws + off); off += (size_t)NN * HS * sizeof(float);
    unsigned short* UfwB  = (unsigned short*)(ws + off); off += (size_t)256 * 256 * 2;
    unsigned short* WcatB = (unsigned short*)(ws + off); off += (size_t)768 * 512 * 2;
    (void)ws_size; (void)in_sizes; (void)n_in; (void)out_size;

    // zero the two f32 accumulators (h_tild, c_agg are contiguous at ws start)
    long n4 = (long)NN * HS * 2 / 4;
    zero_ws_kernel<<<4096, 256, 0, stream>>>((float4*)ws, n4);

    convert_weights_kernel<<<(768 * 512 + 255) / 256, 256, 0, stream>>>(
        U_f_w, W_iou, U_iou, UfwB, WcatB);

    edge_kernel<<<NE / 16, 128, 0, stream>>>(h, c, esrc, edst, UfwB, U_f_b, h_tild, c_agg);

    float* out_h = (float*)d_out;
    float* out_c = out_h + (size_t)NN * HS;
    node_kernel<<<NN / 16, 256, 0, stream>>>(x, h_tild, c_agg, WcatB, b_iou, out_h, out_c);
}
